// GroupedQueryAttention_70798240907760
// MI455X (gfx1250) — matrix operand order repaired and checked by reading, not compiled, not run on hardware
//
#include <hip/hip_runtime.h>

// ---------------------------------------------------------------------------
// GQA (16 heads, 4 KV groups, head_dim 64, T=4096, DIM=1024) on MI455X.
// All matmuls via v_wmma_f32_16x16x32_bf16 (bf16 in, f32 accumulate).
// GEMMs + attention: async global->LDS double-buffered tiles (ASYNCcnt path).
// Attention: flash streaming, DPP-based softmax reductions (no ds_bpermute).
// ---------------------------------------------------------------------------

#define T_SEQ   4096
#define DIM_    1024
#define NHEADS  16
#define NGROUPS 4
#define HDIM    64
#define KVW     (NGROUPS * HDIM)   // 256
#define SCALE_  0.125f             // 64^-0.5

typedef __bf16 bf16_t;
typedef bf16_t       v16bf __attribute__((ext_vector_type(16)));
typedef float        v8f   __attribute__((ext_vector_type(8)));
typedef unsigned int v4u   __attribute__((ext_vector_type(4)));

union Frag {            // 16 bf16 = 32 bytes = two 16-byte quads
  v16bf bf;
  v4u   q[2];
};

__device__ __forceinline__ unsigned short f2bf(float f) {
  unsigned int u = __float_as_uint(f);
  u += 0x7FFFu + ((u >> 16) & 1u);        // round-to-nearest-even
  return (unsigned short)(u >> 16);
}

__device__ __forceinline__ v8f wmma_bf16(const Frag& a, const Frag& b, v8f c) {
  return __builtin_amdgcn_wmma_f32_16x16x32_bf16(
      /*neg_a=*/false, a.bf, /*neg_b=*/false, b.bf,
      /*c_mod=*/(short)0, c, /*reuse_a=*/false, /*reuse_b=*/false);
}

// ------------------- cross-lane reductions via DPP (wave32) ----------------

#if __has_builtin(__builtin_amdgcn_mov_dpp8) && __has_builtin(__builtin_amdgcn_update_dpp)
#define USE_DPP 1
#endif

template <int M>
__device__ __forceinline__ float xor_dpp8(float x) {
#ifdef USE_DPP
  constexpr int sel = (((0 ^ M) & 7) << 0)  | (((1 ^ M) & 7) << 3)  |
                      (((2 ^ M) & 7) << 6)  | (((3 ^ M) & 7) << 9)  |
                      (((4 ^ M) & 7) << 12) | (((5 ^ M) & 7) << 15) |
                      (((6 ^ M) & 7) << 18) | (((7 ^ M) & 7) << 21);
  return __int_as_float(__builtin_amdgcn_mov_dpp8(__float_as_int(x), sel));
#else
  return __shfl_xor(x, M, 32);
#endif
}

__device__ __forceinline__ float xor8_row_ror(float x) {
#ifdef USE_DPP
  // DPP16 ROW_ROR:8 (0x128): lane L <-> L^8 within each 16-lane row
  return __int_as_float(__builtin_amdgcn_update_dpp(
      0, __float_as_int(x), 0x128, 0xf, 0xf, true));
#else
  return __shfl_xor(x, 8, 32);
#endif
}

__device__ __forceinline__ float red_max16(float v) {
  v = fmaxf(v, xor_dpp8<1>(v));
  v = fmaxf(v, xor_dpp8<2>(v));
  v = fmaxf(v, xor_dpp8<4>(v));
  v = fmaxf(v, xor8_row_ror(v));
  return v;
}
__device__ __forceinline__ float red_sum16(float v) {
  v += xor_dpp8<1>(v);
  v += xor_dpp8<2>(v);
  v += xor_dpp8<4>(v);
  v += xor8_row_ror(v);
  return v;
}

// ---------------- CDNA5 async global -> LDS copy (ASYNCcnt) ----------------
// Generic pointers to __shared__ carry the LDS offset in their low 32 bits
// (flat->LDS truncates addr[31:0]), which is exactly what VDST wants.

__device__ __forceinline__ void async_b128(const unsigned short* gptr,
                                           unsigned short* lptr) {
  asm volatile("global_load_async_to_lds_b128 %0, %1, off"
               :: "v"((unsigned)(size_t)lptr), "v"(gptr)
               : "memory");
}
__device__ __forceinline__ void wait_async0() {
  asm volatile("s_wait_asynccnt 0" ::: "memory");
}

// --------------------------- packing kernels -------------------------------

__global__ void cvt_f32_bf16(const float* __restrict__ in,
                             unsigned short* __restrict__ out, int n) {
  int i = blockIdx.x * blockDim.x + threadIdx.x;
  if (i < n) out[i] = f2bf(in[i]);
}

// in: [K][N] f32 row-major -> out: [N][K] bf16.  32x32 LDS-tiled so both the
// f32 reads and the bf16 writes are coalesced. Block = 256 threads (32x8).
__global__ void __launch_bounds__(256)
transpose_pack(const float* __restrict__ in, unsigned short* __restrict__ out,
               int K, int N) {
  __shared__ float tile[32][33];            // +1 to avoid bank conflicts
  const int k0 = blockIdx.x * 32;
  const int n0 = blockIdx.y * 32;
  const int tx = threadIdx.x & 31;
  const int ty = threadIdx.x >> 5;          // 0..7
#pragma unroll
  for (int r = ty; r < 32; r += 8)
    tile[r][tx] = in[(size_t)(k0 + r) * N + n0 + tx];
  __syncthreads();
#pragma unroll
  for (int r = ty; r < 32; r += 8)
    out[(size_t)(n0 + r) * K + k0 + tx] = f2bf(tile[tx][r]);
}

// ------------------------------ WMMA GEMM ----------------------------------
// C[M][N] = A[M][K] * B[K][N], A row-major bf16, B given transposed [N][K].
// Block: 128 threads = 4 waves; block tile 128(M) x 64(N); wave tile 32x64.
// B tile (64n x 32k, 4KB) async-staged to LDS, double-buffered, shared by
// all 4 waves. MODE 0: bf16 row-major | MODE 1: bf16 transposed | MODE 2: f32.
template <int MODE>
__global__ void __launch_bounds__(128)
gemm_bf16(const unsigned short* __restrict__ A,
          const unsigned short* __restrict__ Bt,
          void* __restrict__ Out, int K, int ldout) {
  __shared__ unsigned short bs[2][64 * 32];

  const int lane = threadIdx.x & 31;
  const int wave = threadIdx.x >> 5;
  const int col  = lane & 15;
  const int half = lane >> 4;
  const int m0 = blockIdx.x * 128 + wave * 32;
  const int n0 = blockIdx.y * 64;

  // each of 128 threads async-copies two 16B chunks per stage (4KB total)
  auto issue = [&](int k0, int buf) {
#pragma unroll
    for (int s = 0; s < 2; ++s) {
      int c  = threadIdx.x + s * 128;   // chunk 0..255 (8 bf16 each)
      int n  = c >> 2;
      int kq = (c & 3) * 8;
      async_b128(Bt + (size_t)(n0 + n) * K + k0 + kq,
                 &bs[buf][n * 32 + kq]);
    }
  };

  v8f acc[2][4] = {};
  const unsigned short* arow0 = A + (size_t)(m0 + col) * K + (half ? 8 : 0);
  const unsigned short* arow1 = arow0 + (size_t)16 * K;

  issue(0, 0);
  for (int k0 = 0; k0 < K; k0 += 32) {
    const int buf = (k0 >> 5) & 1;
    wait_async0();          // my writes to bs[buf] done
    __syncthreads();        // everyone's writes done; prior reads of buf^1 done
    if (k0 + 32 < K) issue(k0 + 32, buf ^ 1);

    Frag fa0, fa1;
    fa0.q[0] = *(const v4u*)(arow0 + k0);
    fa0.q[1] = *(const v4u*)(arow0 + k0 + 16);
    fa1.q[0] = *(const v4u*)(arow1 + k0);
    fa1.q[1] = *(const v4u*)(arow1 + k0 + 16);
#pragma unroll
    for (int nt = 0; nt < 4; ++nt) {
      const unsigned short* bp = &bs[buf][(nt * 16 + col) * 32 + half * 8];
      Frag fb;
      fb.q[0] = *(const v4u*)(bp);
      fb.q[1] = *(const v4u*)(bp + 16);
      acc[0][nt] = wmma_bf16(fa0, fb, acc[0][nt]);
      acc[1][nt] = wmma_bf16(fa1, fb, acc[1][nt]);
    }
  }

  // C layout: VGPR j, lane -> row (mb + j), col (n0 + nt*16 + col)
#pragma unroll
  for (int mt = 0; mt < 2; ++mt) {
    const int mb = m0 + mt * 16 + half * 8;
    if (MODE == 0) {
      unsigned short* o = (unsigned short*)Out;
#pragma unroll
      for (int nt = 0; nt < 4; ++nt)
#pragma unroll
        for (int j = 0; j < 8; ++j)
          o[(size_t)(mb + j) * ldout + n0 + nt * 16 + col] =
              f2bf(acc[mt][nt][j]);
    } else if (MODE == 1) {
      unsigned short* o = (unsigned short*)Out;
#pragma unroll
      for (int nt = 0; nt < 4; ++nt)
#pragma unroll
        for (int j = 0; j < 8; ++j)
          o[(size_t)(n0 + nt * 16 + col) * ldout + mb + j] =
              f2bf(acc[mt][nt][j]);
    } else {
      float* o = (float*)Out;
#pragma unroll
      for (int nt = 0; nt < 4; ++nt)
#pragma unroll
        for (int j = 0; j < 8; ++j)
          o[(size_t)(mb + j) * ldout + n0 + nt * 16 + col] = acc[mt][nt][j];
    }
  }
}

// --------------------------- flash attention -------------------------------
// Block = 8 waves, one head; each wave owns 16 query rows. All waves share the
// same KV group, so K (32key x 64d) and V^T (64d x 32t) chunks are async-
// staged into LDS once per block, double-buffered on ASYNCcnt, then every
// fragment read is a contiguous ds_load_b128.
// Qb: [T][1024] ; Kb: [T][256] ; Vt: [256][T] ; Ab: [T][1024]  (all bf16)
__global__ void __launch_bounds__(256)
gqa_flash(const unsigned short* __restrict__ Qb,
          const unsigned short* __restrict__ Kb,
          const unsigned short* __restrict__ Vt,
          unsigned short* __restrict__ Ab) {
  __shared__ unsigned short Ks[2][32 * 64];       // [key][d]   4KB x2
  __shared__ unsigned short Vs[2][64 * 32];       // [d][t]     4KB x2
  __shared__ unsigned short p_lds[8 * 16 * 32];   // per-wave P relayout, 8KB

  const int lane = threadIdx.x & 31;
  const int wave = threadIdx.x >> 5;
  const int col  = lane & 15;
  const int half = lane >> 4;
  const int h  = blockIdx.y;          // head
  const int g  = h >> 2;              // kv group (rep = 4)
  const int q0 = (blockIdx.x * 8 + wave) * 16;

  unsigned short* pw = p_lds + wave * (16 * 32);

  // async-stage one 32-key chunk of K and V^T (8KB): 2 x b128 per thread
  auto issue = [&](int kc, int buf) {
    const int c = threadIdx.x;                 // 0..255
    {
      int key = c >> 3, dq = (c & 7) * 8;      // 32 rows x 8 chunks
      async_b128(Kb + (size_t)(kc + key) * KVW + g * HDIM + dq,
                 &Ks[buf][key * 64 + dq]);
    }
    {
      int d = c >> 2, tq = (c & 3) * 8;        // 64 rows x 4 chunks
      async_b128(Vt + (size_t)(g * HDIM + d) * T_SEQ + kc + tq,
                 &Vs[buf][d * 32 + tq]);
    }
  };

  // Q A-fragments for d chunks [0,32) and [32,64): resident for whole loop
  Frag fq[2];
  {
    const unsigned short* qrow =
        Qb + (size_t)(q0 + col) * DIM_ + h * HDIM + (half ? 8 : 0);
    fq[0].q[0] = *(const v4u*)(qrow);
    fq[0].q[1] = *(const v4u*)(qrow + 16);
    fq[1].q[0] = *(const v4u*)(qrow + 32);
    fq[1].q[1] = *(const v4u*)(qrow + 48);
  }

  v8f o[4] = {};                // O accumulator, 16q x 64d (4 col tiles)
  float m[8], l[8];             // per-row running max / sum (row = half*8+j)
#pragma unroll
  for (int j = 0; j < 8; ++j) { m[j] = -1e30f; l[j] = 0.f; }

  issue(0, 0);
  for (int kc = 0; kc < T_SEQ; kc += 32) {
    const int buf = (kc >> 5) & 1;
    wait_async0();          // my async writes to buf done
    __syncthreads();        // everyone's done; prior reads of buf^1 done
    if (kc + 32 < T_SEQ) issue(kc + 32, buf ^ 1);

    // ---- S = (Q K^T) for 32 keys: two 16-key C tiles, K-dim = HDIM=64 ----
    v8f s0 = {}, s1 = {};
#pragma unroll
    for (int dc = 0; dc < 2; ++dc) {
      const unsigned short* kr0 = &Ks[buf][(col) * 64      + dc * 32 + half * 8];
      const unsigned short* kr1 = &Ks[buf][(16 + col) * 64 + dc * 32 + half * 8];
      Frag fk0, fk1;
      fk0.q[0] = *(const v4u*)(kr0); fk0.q[1] = *(const v4u*)(kr0 + 16);
      fk1.q[0] = *(const v4u*)(kr1); fk1.q[1] = *(const v4u*)(kr1 + 16);
      s0 = wmma_bf16(fq[dc], fk0, s0);
      s1 = wmma_bf16(fq[dc], fk1, s1);
    }

    // ---- online softmax: DPP reductions across the 16 columns ----
    float p0[8], p1[8], sc[8];
#pragma unroll
    for (int j = 0; j < 8; ++j) {
      float sa = s0[j] * SCALE_;
      float sb = s1[j] * SCALE_;
      float t  = red_max16(fmaxf(sa, sb));
      float mn = fmaxf(m[j], t);
      sc[j] = __expf(m[j] - mn);
      m[j]  = mn;
      p0[j] = __expf(sa - mn);
      p1[j] = __expf(sb - mn);
      float rs = red_sum16(p0[j] + p1[j]);
      l[j] = l[j] * sc[j] + rs;
    }
#pragma unroll
    for (int dt = 0; dt < 4; ++dt)
#pragma unroll
      for (int j = 0; j < 8; ++j)
        o[dt][j] = o[dt][j] * sc[j];

    // ---- relayout P: C-layout -> A-layout via per-wave LDS tile ----
#pragma unroll
    for (int j = 0; j < 8; ++j) {
      int row = half * 8 + j;
      pw[row * 32 + col]      = f2bf(p0[j]);
      pw[row * 32 + 16 + col] = f2bf(p1[j]);
    }
    asm volatile("s_wait_dscnt 0" ::: "memory");  // same-wave DS store->load
    Frag fp;
    {
      const unsigned short* pr = pw + col * 32 + (half ? 8 : 0);
      fp.q[0] = *(const v4u*)(pr);
      fp.q[1] = *(const v4u*)(pr + 16);
    }

    // ---- O += P * V : B-operand from LDS V^T tile ----
#pragma unroll
    for (int dt = 0; dt < 4; ++dt) {
      const unsigned short* vr = &Vs[buf][(dt * 16 + col) * 32 + half * 8];
      Frag fv;
      fv.q[0] = *(const v4u*)(vr);
      fv.q[1] = *(const v4u*)(vr + 16);
      o[dt] = wmma_bf16(fp, fv, o[dt]);
    }
  }

  // ---- epilogue: normalize, store bf16 attention output ----
#pragma unroll
  for (int j = 0; j < 8; ++j) l[j] = 1.0f / l[j];
#pragma unroll
  for (int dt = 0; dt < 4; ++dt)
#pragma unroll
    for (int j = 0; j < 8; ++j)
      Ab[(size_t)(q0 + half * 8 + j) * DIM_ + h * HDIM + dt * 16 + col] =
          f2bf(o[dt][j] * l[j]);
}

// ------------------------------- launcher ----------------------------------

extern "C" void kernel_launch(void* const* d_in, const int* in_sizes, int n_in,
                              void* d_out, int out_size, void* d_ws,
                              size_t ws_size, hipStream_t stream) {
  const float* x  = (const float*)d_in[0];
  const float* Wq = (const float*)d_in[1];
  const float* Wk = (const float*)d_in[2];
  const float* Wv = (const float*)d_in[3];
  const float* Wo = (const float*)d_in[4];

  // workspace layout (bf16 stored as ushort); total ~33 MB
  unsigned short* ws = (unsigned short*)d_ws;
  size_t off = 0;
  unsigned short* xb  = ws + off; off += (size_t)T_SEQ * DIM_;  // x  bf16
  unsigned short* Wqt = ws + off; off += (size_t)DIM_ * DIM_;   // Wq^T
  unsigned short* Wkt = ws + off; off += (size_t)KVW * DIM_;    // Wk^T
  unsigned short* Wvt = ws + off; off += (size_t)KVW * DIM_;    // Wv^T
  unsigned short* Wot = ws + off; off += (size_t)DIM_ * DIM_;   // Wo^T
  unsigned short* Qb  = ws + off; off += (size_t)T_SEQ * DIM_;  // Q
  unsigned short* Kb  = ws + off; off += (size_t)T_SEQ * KVW;   // K
  unsigned short* Vt  = ws + off; off += (size_t)KVW * T_SEQ;   // V^T
  unsigned short* Ab  = ws + off; off += (size_t)T_SEQ * DIM_;  // attn out

  const int nx = T_SEQ * DIM_;
  cvt_f32_bf16<<<(nx + 255) / 256, 256, 0, stream>>>(x, xb, nx);
  transpose_pack<<<dim3(DIM_ / 32, DIM_ / 32), 256, 0, stream>>>(Wq, Wqt, DIM_, DIM_);
  transpose_pack<<<dim3(DIM_ / 32, KVW / 32),  256, 0, stream>>>(Wk, Wkt, DIM_, KVW);
  transpose_pack<<<dim3(DIM_ / 32, KVW / 32),  256, 0, stream>>>(Wv, Wvt, DIM_, KVW);
  transpose_pack<<<dim3(DIM_ / 32, DIM_ / 32), 256, 0, stream>>>(Wo, Wot, DIM_, DIM_);

  // Q = x @ Wq   (4096x1024 @ 1024x1024), bf16 out
  gemm_bf16<0><<<dim3(T_SEQ / 128, DIM_ / 64), 128, 0, stream>>>(xb, Wqt, Qb, DIM_, DIM_);
  // K = x @ Wk   (4096x1024 @ 1024x256), bf16 out
  gemm_bf16<0><<<dim3(T_SEQ / 128, KVW / 64), 128, 0, stream>>>(xb, Wkt, Kb, DIM_, KVW);
  // V^T = (x @ Wv)^T  stored [256][4096] for the P*V B-operand
  gemm_bf16<1><<<dim3(T_SEQ / 128, KVW / 64), 128, 0, stream>>>(xb, Wvt, Vt, DIM_, T_SEQ);

  // flash attention: 256 q-tiles of 16 rows x 16 heads, 8 waves/block
  gqa_flash<<<dim3(T_SEQ / 16 / 8, NHEADS), 256, 0, stream>>>(Qb, Kb, Vt, Ab);

  // out = attn @ Wo -> f32 d_out
  gemm_bf16<2><<<dim3(T_SEQ / 128, DIM_ / 64), 128, 0, stream>>>(Ab, Wot, d_out, DIM_, DIM_);
}
